// SingleGraph_76459007803948
// MI455X (gfx1250) — matrix operand order repaired and checked by reading, not compiled, not run on hardware
//
#include <hip/hip_runtime.h>
#include <hip/hip_bf16.h>
#include <math.h>

// ---------------- problem constants (match reference) ----------------
#define BDIM 8
#define SDIM 2048      // power of two: row decompose via shift/mask
#define SSHIFT 11
#define HDIM 768
#define KDIM 128
#define CDIM 256

// ---------------- GEMM tiling ----------------
#define BM 128         // block M tile
#define BN 64          // block N tile
#define BK 32          // K step (one bf16 WMMA depth)
#define LSTR 40        // LDS row stride in bf16 (80 B: 16B-aligned rows, skewed banks)

typedef __bf16 v16bf __attribute__((ext_vector_type(16)));
typedef __bf16 v8bf  __attribute__((ext_vector_type(8)));
typedef float  v8f   __attribute__((ext_vector_type(8)));

__device__ __forceinline__ v16bf combine16(v8bf lo, v8bf hi) {
  return __builtin_shufflevector(lo, hi, 0,1,2,3,4,5,6,7,8,9,10,11,12,13,14,15);
}
__device__ __forceinline__ v8bf pack8(float4 a, float4 b) {
  v8bf r;
  r[0] = (__bf16)a.x; r[1] = (__bf16)a.y; r[2] = (__bf16)a.z; r[3] = (__bf16)a.w;
  r[4] = (__bf16)b.x; r[5] = (__bf16)b.y; r[6] = (__bf16)b.z; r[7] = (__bf16)b.w;
  return r;
}

// =====================================================================
// bf16-WMMA GEMM, C[m, colOff+n] (+opt tanh) = sum_kk A_win[m,kk] * W[n,kk]
// A_win folds the conv im2col window:
//   m -> (b = m>>SSHIFT, s = m&(SDIM-1)); kk -> (t = kk/HIN, h = kk%HIN)
//   A_win[m,kk] = A[b, s + t - pad, h]  (0 outside [0,S))
// HIN is a template constant -> division strength-reduced; staged chunks are
// 16-element aligned and HIN%16==0, so t is uniform per chunk -> one div /
// thread / K-step and full 128-bit global loads + packed bf16 LDS stores.
// Block: 256 thr = 8 waves in 4(M)x2(N); wave tile 32x32 = 4 WMMAs/K-step.
// =====================================================================
template <int HIN, bool TANH>
__global__ __launch_bounds__(256)
void gemm_bf16_wmma(const float* __restrict__ A,
                    const float* __restrict__ W,   // [N, Kd] row-major f32
                    float* __restrict__ C,
                    int pad, int Kd, int ldC, int colOff)
{
  __shared__ __align__(16) __bf16 As[BM * LSTR];
  __shared__ __align__(16) __bf16 Bs[BN * LSTR];

  const int tid  = threadIdx.x;
  const int lane = tid & 31;
  const int wave = tid >> 5;       // 0..7
  const int wm   = wave & 3;       // M 32-row sub-tile (0..3)
  const int wn   = wave >> 2;      // N 32-col half (0..1)
  const int g    = lane >> 4;      // lane group within wave32
  const int l16  = lane & 15;

  const long rowBase = (long)blockIdx.x * BM;
  const int  nBase   = blockIdx.y * BN;

  // A fill: 128 rows x 32 cols; 2 threads/row, 16 halfs each
  const int afr = tid >> 1;
  const int afc = (tid & 1) << 4;
  // B fill: 64 rows x 32 cols; 4 threads/row, 8 halfs each
  const int bfr = tid >> 2;
  const int bfc = (tid & 3) << 3;

  const long am = rowBase + afr;
  const int  ab = (int)(am >> SSHIFT);
  const int  as = (int)(am & (SDIM - 1));
  const float* Abase = A + ((long)ab << SSHIFT) * HIN;

  v8f acc00 = {}, acc01 = {}, acc10 = {}, acc11 = {};

  for (int k0 = 0; k0 < Kd; k0 += BK) {
    // ---- stage A tile: one 16-elem chunk per thread ----
    {
      const int cs = k0 + afc;          // chunk start, multiple of 16
      const int t  = cs / HIN;          // constant division (mul/shift)
      const int h0 = cs - t * HIN;
      const int sr = as + t - pad;
      v8bf p0, p1;
      if (sr >= 0 && sr < SDIM) {
        const float4* src = (const float4*)(Abase + (long)sr * HIN + h0);
        p0 = pack8(src[0], src[1]);
        p1 = pack8(src[2], src[3]);
      } else {
        p0 = (v8bf)(__bf16)0.0f;
        p1 = (v8bf)(__bf16)0.0f;
      }
      *(v8bf*)&As[afr * LSTR + afc]     = p0;
      *(v8bf*)&As[afr * LSTR + afc + 8] = p1;
    }
    // ---- stage B tile: one 8-elem chunk per thread ----
    {
      const float4* src = (const float4*)(W + (long)(nBase + bfr) * Kd + k0 + bfc);
      *(v8bf*)&Bs[bfr * LSTR + bfc] = pack8(src[0], src[1]);
      if (k0 + BK < Kd) __builtin_prefetch(src + BK / 4, 0, 1);
    }
    __syncthreads();

    // ---- fragments per ISA 7.12.2 16-bit layouts ----
    const __bf16* ap0 = &As[(wm * 32 + l16) * LSTR + (g << 3)];
    const v16bf af0 = combine16(*(const v8bf*)ap0, *(const v8bf*)(ap0 + 16));
    const __bf16* ap1 = &As[(wm * 32 + 16 + l16) * LSTR + (g << 3)];
    const v16bf af1 = combine16(*(const v8bf*)ap1, *(const v8bf*)(ap1 + 16));
    const __bf16* bp0 = &Bs[(wn * 32 + l16) * LSTR + (g << 3)];
    const v16bf bf0 = combine16(*(const v8bf*)bp0, *(const v8bf*)(bp0 + 16));
    const __bf16* bp1 = &Bs[(wn * 32 + 16 + l16) * LSTR + (g << 3)];
    const v16bf bf1 = combine16(*(const v8bf*)bp1, *(const v8bf*)(bp1 + 16));

    acc00 = __builtin_amdgcn_wmma_f32_16x16x32_bf16(false, af0, false, bf0, (short)0, acc00, false, false);
    acc01 = __builtin_amdgcn_wmma_f32_16x16x32_bf16(false, af0, false, bf1, (short)0, acc01, false, false);
    acc10 = __builtin_amdgcn_wmma_f32_16x16x32_bf16(false, af1, false, bf0, (short)0, acc10, false, false);
    acc11 = __builtin_amdgcn_wmma_f32_16x16x32_bf16(false, af1, false, bf1, (short)0, acc11, false, false);
    __syncthreads();
  }

  // ---- store: C/D layout lane L: N=L%16, VGPR r: M=r+8*(L/16) ----
  const int mOff0 = wm * 32 + (g << 3);
  const int nCol  = nBase + wn * 32 + l16;
#pragma unroll
  for (int r = 0; r < 8; ++r) {
    const long m0 = rowBase + mOff0 + r;
    const long m1 = m0 + 16;
    float a = acc00[r], b = acc01[r], c = acc10[r], d = acc11[r];
    if (TANH) { a = tanhf(a); b = tanhf(b); c = tanhf(c); d = tanhf(d); }
    C[m0 * ldC + colOff + nCol]      = a;
    C[m0 * ldC + colOff + nCol + 16] = b;
    C[m1 * ldC + colOff + nCol]      = c;
    C[m1 * ldC + colOff + nCol + 16] = d;
  }
}

// =====================================================================
// conv weight transpose: w[k,H,C] (C innermost) -> out[C, k*H]
// =====================================================================
__global__ void conv_w_transpose(const float* __restrict__ w,
                                 float* __restrict__ out, int kH, int C) {
  const int idx = blockIdx.x * blockDim.x + threadIdx.x;
  if (idx >= kH * C) return;
  const int c  = idx / kH;
  const int kk = idx - c * kH;
  out[idx] = w[(long)kk * C + c];
}

// =====================================================================
// residual add + LayerNorm over H=768 (one block per row, 3 elems/thread)
// =====================================================================
__global__ __launch_bounds__(256)
void add_layernorm(const float* __restrict__ x, const float* __restrict__ resid,
                   const float* __restrict__ gw, const float* __restrict__ bw,
                   float* __restrict__ out) {
  __shared__ float red[256];
  const long base = (long)blockIdx.x * HDIM;
  const int tid = threadIdx.x;
  float v0 = x[base + tid]       + resid[base + tid];
  float v1 = x[base + tid + 256] + resid[base + tid + 256];
  float v2 = x[base + tid + 512] + resid[base + tid + 512];
  red[tid] = v0 + v1 + v2;
  __syncthreads();
  for (int s = 128; s > 0; s >>= 1) { if (tid < s) red[tid] += red[tid + s]; __syncthreads(); }
  const float mu = red[0] * (1.0f / HDIM);
  __syncthreads();
  const float d0 = v0 - mu, d1 = v1 - mu, d2 = v2 - mu;
  red[tid] = d0 * d0 + d1 * d1 + d2 * d2;
  __syncthreads();
  for (int s = 128; s > 0; s >>= 1) { if (tid < s) red[tid] += red[tid + s]; __syncthreads(); }
  const float inv = rsqrtf(red[0] * (1.0f / HDIM) + 1e-5f);
  out[base + tid]       = d0 * inv * gw[tid]       + bw[tid];
  out[base + tid + 256] = d1 * inv * gw[tid + 256] + bw[tid + 256];
  out[base + tid + 512] = d2 * inv * gw[tid + 512] + bw[tid + 512];
}

// =====================================================================
// L2-normalize rows of location [B*S, 128]; one wave32 per row
// =====================================================================
__global__ __launch_bounds__(256)
void normalize_rows(const float* __restrict__ loc, float* __restrict__ normed) {
  const int tid  = threadIdx.x;
  const int lane = tid & 31;
  const long row = (long)blockIdx.x * 8 + (tid >> 5);
  const float4 v = ((const float4*)(loc + row * KDIM))[lane];
  float s = v.x * v.x + v.y * v.y + v.z * v.z + v.w * v.w;
  for (int off = 16; off > 0; off >>= 1) s += __shfl_xor(s, off, 32);
  const float scale = 1.0f / fmaxf(sqrtf(s), 1e-8f);
  const float4 o = make_float4(v.x * scale, v.y * scale, v.z * scale, v.w * scale);
  ((float4*)(normed + row * KDIM))[lane] = o;
}

// =====================================================================
// max_act_len = (int)max_b sum_s mask[b,s]
// =====================================================================
__global__ __launch_bounds__(256)
void compute_maxlen(const float* __restrict__ mask, int* __restrict__ out) {
  __shared__ float red[256];
  const int tid = threadIdx.x;
  float best = 0.0f;
  for (int b = 0; b < BDIM; ++b) {
    float s = 0.0f;
    for (int i = tid; i < SDIM; i += 256) s += mask[(long)b * SDIM + i];
    red[tid] = s;
    __syncthreads();
    for (int t = 128; t > 0; t >>= 1) { if (tid < t) red[tid] += red[tid + t]; __syncthreads(); }
    if (red[0] > best) best = red[0];
    __syncthreads();
  }
  if (tid == 0) out[0] = (int)best;
}

// =====================================================================
// banded pairwise cosine: sim = normed_b @ normed_b^T (WMMA), then
// dis = band ? sim : 0 ; tags = band ? mask_i*mask_j : 0.
// Tiles fully outside the strict upper band take a zero-fill fast path.
// 64x64 tile, 8 waves of 16x32 (2 WMMAs each); output-BW bound anyway.
// =====================================================================
__global__ __launch_bounds__(256)
void banded_cosine_sim(const float* __restrict__ normed,
                       const float* __restrict__ mask,
                       const int* __restrict__ maxlenp,
                       float* __restrict__ dis, float* __restrict__ tags)
{
  const int b  = blockIdx.z;
  const int i0 = blockIdx.x * 64;
  const int j0 = blockIdx.y * 64;
  const int L  = maxlenp[0];
  const int tid = threadIdx.x;

  if ((j0 + 63 <= i0) || (j0 - (i0 + 63) > L)) {
    const long base = ((long)b * SDIM + i0) * SDIM + j0;
#pragma unroll
    for (int r = 0; r < 16; ++r) {
      const int lin = r * 256 + tid;
      const int ii = lin >> 6, jj = lin & 63;
      dis [base + (long)ii * SDIM + jj] = 0.0f;
      tags[base + (long)ii * SDIM + jj] = 0.0f;
    }
    return;
  }

  __shared__ __align__(16) __bf16 As[64 * LSTR];
  __shared__ __align__(16) __bf16 Bs[64 * LSTR];

  const int lane = tid & 31;
  const int wave = tid >> 5;
  const int wm = wave & 3, wn = wave >> 2;
  const int g = lane >> 4, l16 = lane & 15;
  const int fr = tid >> 2, fc = (tid & 3) << 3;

  const float* Ab = normed + (long)b * SDIM * KDIM;
  v8f acc0 = {}, acc1 = {};

  for (int k0 = 0; k0 < KDIM; k0 += BK) {
    {
      const float4* sa = (const float4*)(Ab + (long)(i0 + fr) * KDIM + k0 + fc);
      const float4* sb = (const float4*)(Ab + (long)(j0 + fr) * KDIM + k0 + fc);
      *(v8bf*)&As[fr * LSTR + fc] = pack8(sa[0], sa[1]);
      *(v8bf*)&Bs[fr * LSTR + fc] = pack8(sb[0], sb[1]);
    }
    __syncthreads();

    const __bf16* ap = &As[(wm * 16 + l16) * LSTR + (g << 3)];
    const v16bf af = combine16(*(const v8bf*)ap, *(const v8bf*)(ap + 16));
    const __bf16* bp0 = &Bs[(wn * 32 + l16) * LSTR + (g << 3)];
    const v16bf bf0 = combine16(*(const v8bf*)bp0, *(const v8bf*)(bp0 + 16));
    const __bf16* bp1 = &Bs[(wn * 32 + 16 + l16) * LSTR + (g << 3)];
    const v16bf bf1 = combine16(*(const v8bf*)bp1, *(const v8bf*)(bp1 + 16));

    acc0 = __builtin_amdgcn_wmma_f32_16x16x32_bf16(false, af, false, bf0, (short)0, acc0, false, false);
    acc1 = __builtin_amdgcn_wmma_f32_16x16x32_bf16(false, af, false, bf1, (short)0, acc1, false, false);
    __syncthreads();
  }

  const int mOff = wm * 16 + (g << 3);
  const int jc = j0 + wn * 32 + l16;
  const float mj0 = mask[(long)b * SDIM + jc];
  const float mj1 = mask[(long)b * SDIM + jc + 16];
#pragma unroll
  for (int r = 0; r < 8; ++r) {
    const int i = i0 + mOff + r;
    const float mi = mask[(long)b * SDIM + i];
    const long rowbase = ((long)b * SDIM + i) * SDIM;
    const int dj0 = jc - i, dj1 = jc + 16 - i;
    const bool in0 = (dj0 >= 1) && (dj0 <= L);
    const bool in1 = (dj1 >= 1) && (dj1 <= L);
    dis [rowbase + jc]      = in0 ? acc0[r] : 0.0f;
    dis [rowbase + jc + 16] = in1 ? acc1[r] : 0.0f;
    tags[rowbase + jc]      = in0 ? (mi * mj0) : 0.0f;
    tags[rowbase + jc + 16] = in1 ? (mi * mj1) : 0.0f;
  }
}

// =====================================================================
// host-side orchestration
// =====================================================================
extern "C" void kernel_launch(void* const* d_in, const int* in_sizes, int n_in,
                              void* d_out, int out_size, void* d_ws, size_t ws_size,
                              hipStream_t stream) {
  (void)in_sizes; (void)n_in; (void)out_size; (void)ws_size;

  const float* rep     = (const float*)d_in[0];
  const float* smask   = (const float*)d_in[1];
  const float* ew1     = (const float*)d_in[2];
  const float* ew3     = (const float*)d_in[3];
  const float* ew5     = (const float*)d_in[4];
  const float* lng     = (const float*)d_in[5];
  const float* lnb     = (const float*)d_in[6];
  const float* enc_lin = (const float*)d_in[7];
  const float* enc_tr  = (const float*)d_in[8];
  const float* dec_tr  = (const float*)d_in[9];
  const float* dw1     = (const float*)d_in[10];
  const float* dw3     = (const float*)d_in[11];
  const float* dw5     = (const float*)d_in[12];
  const float* dec_lin = (const float*)d_in[13];
  const float* out_lin = (const float*)d_in[14];

  float* out   = (float*)d_out;
  float* o_loc = out;                                        // [B,S,K]
  float* o_dis = o_loc + (size_t)BDIM * SDIM * KDIM;         // [B,S,S]
  float* o_tag = o_dis + (size_t)BDIM * SDIM * SDIM;         // [B,S,S]
  float* o_dec = o_tag + (size_t)BDIM * SDIM * SDIM;         // [B,S,H]

  float* ws = (float*)d_ws;
  int*   d_maxlen = (int*)d_ws;             // 4 B at base; 16 floats reserved
  size_t off = 16;
  float* buf1   = ws + off;  off += (size_t)BDIM * SDIM * HDIM;
  float* buf2   = ws + off;  off += (size_t)BDIM * SDIM * HDIM;
  float* normed = ws + off;  off += (size_t)BDIM * SDIM * KDIM;
  float* ew1t = ws + off;    off += (size_t)1 * HDIM * CDIM;
  float* ew3t = ws + off;    off += (size_t)3 * HDIM * CDIM;
  float* ew5t = ws + off;    off += (size_t)5 * HDIM * CDIM;
  float* dw1t = ws + off;    off += (size_t)1 * HDIM * CDIM;
  float* dw3t = ws + off;    off += (size_t)3 * HDIM * CDIM;
  float* dw5t = ws + off;    off += (size_t)5 * HDIM * CDIM;

  const int MT = (BDIM * SDIM) / BM;   // 128 row tiles

  {
    int n1 = 1 * HDIM * CDIM, n3 = 3 * HDIM * CDIM, n5 = 5 * HDIM * CDIM;
    conv_w_transpose<<<(n1 + 255) / 256, 256, 0, stream>>>(ew1, ew1t, 1 * HDIM, CDIM);
    conv_w_transpose<<<(n3 + 255) / 256, 256, 0, stream>>>(ew3, ew3t, 3 * HDIM, CDIM);
    conv_w_transpose<<<(n5 + 255) / 256, 256, 0, stream>>>(ew5, ew5t, 5 * HDIM, CDIM);
    conv_w_transpose<<<(n1 + 255) / 256, 256, 0, stream>>>(dw1, dw1t, 1 * HDIM, CDIM);
    conv_w_transpose<<<(n3 + 255) / 256, 256, 0, stream>>>(dw3, dw3t, 3 * HDIM, CDIM);
    conv_w_transpose<<<(n5 + 255) / 256, 256, 0, stream>>>(dw5, dw5t, 5 * HDIM, CDIM);
  }
  compute_maxlen<<<1, 256, 0, stream>>>(smask, d_maxlen);

  // encoder multi-conv (k=1,3,5) -> buf1[:, 0:256 | 256:512 | 512:768]
  gemm_bf16_wmma<HDIM, false><<<dim3(MT, CDIM / BN), 256, 0, stream>>>(rep, ew1t, buf1, 0, 1 * HDIM, HDIM, 0 * CDIM);
  gemm_bf16_wmma<HDIM, false><<<dim3(MT, CDIM / BN), 256, 0, stream>>>(rep, ew3t, buf1, 1, 3 * HDIM, HDIM, 1 * CDIM);
  gemm_bf16_wmma<HDIM, false><<<dim3(MT, CDIM / BN), 256, 0, stream>>>(rep, ew5t, buf1, 2, 5 * HDIM, HDIM, 2 * CDIM);

  // LN(conv + residual) -> buf2
  add_layernorm<<<BDIM * SDIM, 256, 0, stream>>>(buf1, rep, lng, lnb, buf2);

  // h = tanh(buf2 @ enc_lin^T) -> buf1
  gemm_bf16_wmma<HDIM, true ><<<dim3(MT, HDIM / BN), 256, 0, stream>>>(buf2, enc_lin, buf1, 0, HDIM, HDIM, 0);
  // location = buf1 @ enc_tr^T -> d_out
  gemm_bf16_wmma<HDIM, false><<<dim3(MT, KDIM / BN), 256, 0, stream>>>(buf1, enc_tr, o_loc, 0, HDIM, KDIM, 0);
  // d = location @ dec_tr^T -> buf2   (A feature dim = K = 128)
  gemm_bf16_wmma<KDIM, false><<<dim3(MT, HDIM / BN), 256, 0, stream>>>(o_loc, dec_tr, buf2, 0, KDIM, HDIM, 0);

  // decoder multi-conv -> buf1
  gemm_bf16_wmma<HDIM, false><<<dim3(MT, CDIM / BN), 256, 0, stream>>>(buf2, dw1t, buf1, 0, 1 * HDIM, HDIM, 0 * CDIM);
  gemm_bf16_wmma<HDIM, false><<<dim3(MT, CDIM / BN), 256, 0, stream>>>(buf2, dw3t, buf1, 1, 3 * HDIM, HDIM, 1 * CDIM);
  gemm_bf16_wmma<HDIM, false><<<dim3(MT, CDIM / BN), 256, 0, stream>>>(buf2, dw5t, buf1, 2, 5 * HDIM, HDIM, 2 * CDIM);

  // d = tanh(buf1 @ dec_lin^T) -> buf2 ; decode_out = buf2 @ out_lin^T -> d_out
  gemm_bf16_wmma<HDIM, true ><<<dim3(MT, HDIM / BN), 256, 0, stream>>>(buf1, dec_lin, buf2, 0, HDIM, HDIM, 0);
  gemm_bf16_wmma<HDIM, false><<<dim3(MT, HDIM / BN), 256, 0, stream>>>(buf2, out_lin, o_dec, 0, HDIM, HDIM, 0);

  // cosine band
  normalize_rows<<<(BDIM * SDIM) / 8, 256, 0, stream>>>(o_loc, normed);
  banded_cosine_sim<<<dim3(SDIM / 64, SDIM / 64, BDIM), 256, 0, stream>>>(
      normed, smask, d_maxlen, o_dis, o_tag);
}
